// MolecularInspiredGNN_77378130805146
// MI455X (gfx1250) — compile-verified, matrix operand order checked by reading, not run on hardware
//
#include <hip/hip_runtime.h>
#include <hip/hip_bf16.h>

typedef __attribute__((ext_vector_type(16))) __bf16 v16bf;
typedef __attribute__((ext_vector_type(8)))  float  v8f;
typedef __attribute__((ext_vector_type(4)))  unsigned int v4u;
typedef __attribute__((ext_vector_type(8)))  int    v8i;
typedef __attribute__((ext_vector_type(4)))  int    v4i;

#define NUM_GRAPHS 64

#if defined(__has_builtin)
#if __has_builtin(__builtin_amdgcn_tensor_load_to_lds) && \
    __has_builtin(__builtin_amdgcn_s_wait_tensorcnt)
#define USE_TDM 1
#endif
#endif

// ---------------------------------------------------------------------------
// WMMA helpers (CDNA5 wave32, v_wmma_f32_16x16x32_bf16)
// ---------------------------------------------------------------------------
__device__ __forceinline__ v8f wmma_bf16(v16bf a, v16bf b, v8f c) {
  return __builtin_amdgcn_wmma_f32_16x16x32_bf16(false, a, false, b, (short)0, c,
                                                 false, false);
}

// K index inside a 16-bit A/B fragment: VGPR j holds elements {2j, 2j+1};
// lanes 0-15 (h=0) cover K base, lanes 16-31 (h=1) cover K base + 8.
__device__ __forceinline__ int kidx(int j, int h) {
  return ((j < 4) ? (2 * j) : (16 + 2 * (j - 4))) + 8 * h;
}

// Load a pre-swizzled B fragment: layout [frag][lane32][16 bf16] (32B/lane)
__device__ __forceinline__ v16bf load_bfrag(const __bf16* __restrict__ base,
                                            int frag, int lane) {
  return *(const v16bf*)(base + (size_t)frag * 512 + (size_t)lane * 16);
}

// Build an A fragment (16 rows x 32 K) from an f32 LDS tile with row stride S.
__device__ __forceinline__ v16bf build_afrag_lds(const float* tile, int stride,
                                                 int kc, int lane) {
  int m = lane & 15, h = lane >> 4;
  const float* rowp = tile + m * stride;
  v16bf a;
#pragma unroll
  for (int j = 0; j < 8; ++j) {
    int k = kc * 32 + kidx(j, h);
    a[2 * j]     = (__bf16)rowp[k];
    a[2 * j + 1] = (__bf16)rowp[k + 1];
  }
  return a;
}

// ---------------------------------------------------------------------------
// Weight pre-swizzle: row-major f32 (K x N) -> bf16 B-fragments
// out layout: [kc][nt][lane][16] ; frag = kc*ntiles + nt ; K padded to kc*32
// ---------------------------------------------------------------------------
__global__ void swizzle_weight_kernel(const float* __restrict__ W,
                                      __bf16* __restrict__ out, int K, int N) {
  int kchunks = (K + 31) / 32;
  int ntiles  = N / 16;
  int tid = blockIdx.x * blockDim.x + threadIdx.x;
  int total = kchunks * ntiles * 32;
  if (tid >= total) return;
  int lane = tid & 31;
  int frag = tid >> 5;
  int nt = frag % ntiles;
  int kc = frag / ntiles;
  int h = lane >> 4;
  int n = nt * 16 + (lane & 15);
  __bf16* dst = out + (size_t)frag * 512 + (size_t)lane * 16;
#pragma unroll
  for (int j = 0; j < 8; ++j) {
    int k0 = kc * 32 + kidx(j, 0) + 8 * h;
    float f0 = (k0 < K)     ? W[(size_t)k0 * N + n]       : 0.f;
    float f1 = (k0 + 1 < K) ? W[(size_t)(k0 + 1) * N + n] : 0.f;
    dst[2 * j]     = (__bf16)f0;
    dst[2 * j + 1] = (__bf16)f1;
  }
}

// ---------------------------------------------------------------------------
// Node MLP: enhanced[:,0:128] = relu(x @ w1 + b1) @ w2 + b2
// 128 threads = 4 waves; wave handles a 16-row tile.
// ---------------------------------------------------------------------------
__global__ void __launch_bounds__(128) node_mlp_kernel(
    const float* __restrict__ x, const float* __restrict__ b1v,
    const float* __restrict__ b2v, const __bf16* __restrict__ w1f,
    const __bf16* __restrict__ w2f, float* __restrict__ enhanced, int N) {
  __shared__ float hid[4][16][128];
  int wave = threadIdx.x >> 5, lane = threadIdx.x & 31;
  int m = lane & 15, h = lane >> 4;
  int rowBase = (blockIdx.x * 4 + wave) * 16;
  int row = min(rowBase + m, N - 1);

  // layer 1 A fragment straight from global (K=16 zero-padded to 32)
  v16bf a1;
#pragma unroll
  for (int j = 0; j < 8; ++j) {
    int k = kidx(j, h);
    float f0 = (k < 16)     ? x[(size_t)row * 16 + k]     : 0.f;
    float f1 = (k + 1 < 16) ? x[(size_t)row * 16 + k + 1] : 0.f;
    a1[2 * j]     = (__bf16)f0;
    a1[2 * j + 1] = (__bf16)f1;
  }
#pragma unroll
  for (int nt = 0; nt < 8; ++nt) {
    v8f c = {};
    c = wmma_bf16(a1, load_bfrag(w1f, nt, lane), c);
    int n = nt * 16 + m;
    float bias = b1v[n];
#pragma unroll
    for (int v = 0; v < 8; ++v)
      hid[wave][v + 8 * h][n] = fmaxf(c[v] + bias, 0.f);
  }
  __syncthreads();

  v16bf a2[4];
#pragma unroll
  for (int kc = 0; kc < 4; ++kc)
    a2[kc] = build_afrag_lds(&hid[wave][0][0], 128, kc, lane);

#pragma unroll
  for (int nt = 0; nt < 8; ++nt) {
    v8f c = {};
#pragma unroll
    for (int kc = 0; kc < 4; ++kc)
      c = wmma_bf16(a2[kc], load_bfrag(w2f, kc * 8 + nt, lane), c);
    int n = nt * 16 + m;
    float bias = b2v[n];
#pragma unroll
    for (int v = 0; v < 8; ++v) {
      int r = rowBase + v + 8 * h;
      if (r < N) enhanced[(size_t)r * 192 + n] = c[v] + bias;
    }
  }
}

// ---------------------------------------------------------------------------
// Edge kernel: dist/angle/raw MLPs + e2n projection, atomic scatter into
// enhanced[:,128:192]. 256 threads = 8 waves, wave handles 16 edges.
// ---------------------------------------------------------------------------
__global__ void __launch_bounds__(256) edge_feature_kernel(
    const float* __restrict__ x, const float* __restrict__ edge_attr,
    const int* __restrict__ edge_index,
    const float* __restrict__ dw1, const float* __restrict__ db1,
    const float* __restrict__ db2,
    const float* __restrict__ aw1, const float* __restrict__ ab1,
    const float* __restrict__ ab2,
    const float* __restrict__ rw1, const float* __restrict__ rb1,
    const float* __restrict__ rb2,
    const __bf16* __restrict__ dW2, const __bf16* __restrict__ aW2,
    const __bf16* __restrict__ rW2, const __bf16* __restrict__ e2nW,
    const float* __restrict__ e2nb, float* __restrict__ enhanced, int E) {
  __shared__ float feat[8][16][96];
  __shared__ int scol[8][16];
  int wave = threadIdx.x >> 5, lane = threadIdx.x & 31;
  int m = lane & 15, h = lane >> 4;
  long e0 = ((long)blockIdx.x * 8 + wave) * 16;
  long e = e0 + m;
  bool ev = e < (long)E;
  long ec = ev ? e : (long)E - 1;
  int nr = edge_index[ec];
  int nc = edge_index[(long)E + ec];
  if (h == 0) scol[wave][m] = ev ? nc : -1;

  const float* xr = x + (size_t)nr * 16;
  const float* xc = x + (size_t)nc * 16;
  float rx = xr[0], ry = xr[1], rz = xr[2];
  float sx = xc[0], sy = xc[1], sz = xc[2];
  float dx = rx - sx, dy = ry - sy, dz = rz - sz;
  float dist = sqrtf(dx * dx + dy * dy + dz * dz);
  float dot = rx * sx + ry * sy + rz * sz;
  float ni = sqrtf(rx * rx + ry * ry + rz * rz);
  float nj = sqrtf(sx * sx + sy * sy + sz * sz);
  float ca = dot / (ni * nj + 1e-8f);
  ca = fminf(1.f, fmaxf(-1.f, ca));
  float ang = acosf(ca);
  float ea[8];
#pragma unroll
  for (int t = 0; t < 8; ++t) ea[t] = edge_attr[ec * 8 + t];

  // layer-1 outputs built directly in A-fragment layout (rank-1 MLPs)
  v16bf aD, aA, aR;
#pragma unroll
  for (int j = 0; j < 8; ++j) {
    int k = kidx(j, h);
#pragma unroll
    for (int s = 0; s < 2; ++s) {
      int kk = k + s;
      float hD = fmaxf(dist * dw1[kk] + db1[kk], 0.f);
      float hA = fmaxf(ang * aw1[kk] + ab1[kk], 0.f);
      float hR = rb1[kk];
#pragma unroll
      for (int t = 0; t < 8; ++t) hR += ea[t] * rw1[t * 32 + kk];
      hR = fmaxf(hR, 0.f);
      aD[2 * j + s] = (__bf16)hD;
      aA[2 * j + s] = (__bf16)hA;
      aR[2 * j + s] = (__bf16)hR;
    }
  }
  // layer 2 of each small MLP (32x32) -> feature tile 16x96 in LDS
#pragma unroll
  for (int nt = 0; nt < 2; ++nt) {
    int n = nt * 16 + m;
    v8f cD = {}; cD = wmma_bf16(aD, load_bfrag(dW2, nt, lane), cD);
    v8f cA = {}; cA = wmma_bf16(aA, load_bfrag(aW2, nt, lane), cA);
    v8f cR = {}; cR = wmma_bf16(aR, load_bfrag(rW2, nt, lane), cR);
    float bD = db2[n], bA = ab2[n], bR = rb2[n];
#pragma unroll
    for (int v = 0; v < 8; ++v) {
      int mr = v + 8 * h;
      feat[wave][mr][n]      = cD[v] + bD;
      feat[wave][mr][32 + n] = cA[v] + bA;
      feat[wave][mr][64 + n] = cR[v] + bR;
    }
  }
  __syncthreads();

  // e2n: (16x96) @ (96x64)
  v16bf af[3];
#pragma unroll
  for (int kc = 0; kc < 3; ++kc)
    af[kc] = build_afrag_lds(&feat[wave][0][0], 96, kc, lane);
#pragma unroll
  for (int nt = 0; nt < 4; ++nt) {
    v8f acc = {};
#pragma unroll
    for (int kc = 0; kc < 3; ++kc)
      acc = wmma_bf16(af[kc], load_bfrag(e2nW, kc * 4 + nt, lane), acc);
    int n = nt * 16 + m;
    float bias = e2nb[n];
#pragma unroll
    for (int v = 0; v < 8; ++v) {
      int cidx = scol[wave][v + 8 * h];
      if (cidx >= 0)
        atomicAdd(&enhanced[(size_t)cidx * 192 + 128 + n], acc[v] + bias);
    }
  }
}

// ---------------------------------------------------------------------------
// neigh scatter: gin_in[col] += enhanced[row]  (gin_in pre-seeded = enhanced)
// thread per (edge, float4 chunk of 192)
// ---------------------------------------------------------------------------
__global__ void scatter_neigh_kernel(const float* __restrict__ enhanced,
                                     float* __restrict__ gin_in,
                                     const int* __restrict__ edge_index, int E) {
  long tid = (long)blockIdx.x * blockDim.x + threadIdx.x;
  long total = (long)E * 48;
  if (tid >= total) return;
  long e = tid / 48;
  int q = (int)(tid % 48);
  int r = edge_index[e];
  int c = edge_index[(long)E + e];
  float4 v = ((const float4*)(enhanced + (size_t)r * 192))[q];
  float* dst = gin_in + (size_t)c * 192 + (size_t)q * 4;
  atomicAdd(dst + 0, v.x);
  atomicAdd(dst + 1, v.y);
  atomicAdd(dst + 2, v.z);
  atomicAdd(dst + 3, v.w);
}

__global__ void counts_kernel(const int* __restrict__ batch,
                              float* __restrict__ counts, int N) {
  int i = blockIdx.x * blockDim.x + threadIdx.x;
  if (i < N) atomicAdd(&counts[batch[i]], 1.0f);
}

// ---------------------------------------------------------------------------
// GIN MLP + fused graph pooling:
//   out = relu(relu(gin_in @ w1 + b1) @ w2 + b2); sums[batch] += out
// Input tile staged HBM->LDS by the Tensor Data Mover when available.
// ---------------------------------------------------------------------------
__global__ void __launch_bounds__(128) gin_mlp_kernel(
    const float* __restrict__ gin_in, const float* __restrict__ b1v,
    const float* __restrict__ b2v, const __bf16* __restrict__ w1f,
    const __bf16* __restrict__ w2f, const int* __restrict__ batch,
    float* __restrict__ sums, int N) {
  __shared__ float tile[4][16][192];
  int wave = threadIdx.x >> 5, lane = threadIdx.x & 31;
  int m = lane & 15, h = lane >> 4;
  int rowBase = (blockIdx.x * 4 + wave) * 16;

#ifdef USE_TDM
  {
    // TDM 2D descriptor: tile = 192 x 16 f32, row stride 192, OOB rows -> 0
    int rowsValid = N - rowBase;
    rowsValid = rowsValid < 0 ? 0 : (rowsValid > 16 ? 16 : rowsValid);
    unsigned int ldsAddr =
        (unsigned int)(unsigned long long)(uintptr_t)(&tile[wave][0][0]);
    unsigned long long gaddr =
        (unsigned long long)(uintptr_t)(gin_in + (size_t)rowBase * 192);
    // Group0: count=1 | lds_addr | global_addr | type=2
    v4u g0 = {1u, ldsAddr, (unsigned int)gaddr,
              (unsigned int)(gaddr >> 32) | (2u << 30)};
    // Group1: data_size=4B; tensor_dim0=192; tensor_dim1=rowsValid;
    //         tile_dim0=192; tile_dim1=16; tensor_dim0_stride=192
    v8i g1 = {(int)(2u << 16),               // data_size = 2 (4 bytes)
              (int)(192u << 16),             // tensor_dim0 lo16 @ bits[63:48]
              (int)(((unsigned)rowsValid) << 16),  // tensor_dim1 lo16 @ [95:80]
              (int)(192u << 16),             // tile_dim0 @ bits[127:112]
              16,                            // tile_dim1 @ bits[143:128]
              192,                           // tensor_dim0_stride @ [191:160]
              0, 0};
    v4i gz = {0, 0, 0, 0};
#if __clang_major__ >= 23
    v8i gz8 = {0, 0, 0, 0, 0, 0, 0, 0};
    __builtin_amdgcn_tensor_load_to_lds(g0, g1, gz, gz, gz8, 0);
#else
    __builtin_amdgcn_tensor_load_to_lds(g0, g1, gz, gz, 0);
#endif
    __builtin_amdgcn_s_wait_tensorcnt(0);
  }
#else
  // fallback: cooperative per-wave tile load (float4)
  for (int i = lane; i < 16 * 48; i += 32) {
    int rr = i / 48, qq = i % 48;
    int grow = min(rowBase + rr, N - 1);
    ((float4*)&tile[wave][rr][0])[qq] =
        ((const float4*)(gin_in + (size_t)grow * 192))[qq];
  }
#endif
  __syncthreads();

  v16bf a1[6];
#pragma unroll
  for (int kc = 0; kc < 6; ++kc)
    a1[kc] = build_afrag_lds(&tile[wave][0][0], 192, kc, lane);

  // layer 1: hidden tile written back into same LDS buffer (cols 0..127)
#pragma unroll
  for (int nt = 0; nt < 8; ++nt) {
    v8f c = {};
#pragma unroll
    for (int kc = 0; kc < 6; ++kc)
      c = wmma_bf16(a1[kc], load_bfrag(w1f, kc * 8 + nt, lane), c);
    int n = nt * 16 + m;
    float bias = b1v[n];
#pragma unroll
    for (int v = 0; v < 8; ++v)
      tile[wave][v + 8 * h][n] = fmaxf(c[v] + bias, 0.f);
  }
  __syncthreads();

  v16bf a2[4];
#pragma unroll
  for (int kc = 0; kc < 4; ++kc)
    a2[kc] = build_afrag_lds(&tile[wave][0][0], 192, kc, lane);

#pragma unroll
  for (int nt = 0; nt < 8; ++nt) {
    v8f c = {};
#pragma unroll
    for (int kc = 0; kc < 4; ++kc)
      c = wmma_bf16(a2[kc], load_bfrag(w2f, kc * 8 + nt, lane), c);
    int n = nt * 16 + m;
    float bias = b2v[n];
#pragma unroll
    for (int v = 0; v < 8; ++v) {
      int r = rowBase + v + 8 * h;
      if (r < N) {
        float val = fmaxf(c[v] + bias, 0.f);
        atomicAdd(&sums[(size_t)batch[r] * 128 + n], val);
      }
    }
  }
}

// ---------------------------------------------------------------------------
// Final head: graph_emb = sums/max(counts,1); out = relu(ge@fc1+b1)@fc2+b2
// Single block; out pre-zeroed, accumulated with atomics.
// ---------------------------------------------------------------------------
__global__ void __launch_bounds__(128) final_kernel(
    const float* __restrict__ sums, const float* __restrict__ counts,
    const float* __restrict__ fw1, const float* __restrict__ fb1,
    const float* __restrict__ fw2, const float* __restrict__ fb2,
    float* __restrict__ out) {
  __shared__ float ge[NUM_GRAPHS * 128];
  int t = threadIdx.x;
  for (int i = t; i < NUM_GRAPHS * 128; i += 128) {
    int g = i >> 7;
    ge[i] = sums[i] / fmaxf(counts[g], 1.0f);
  }
  __syncthreads();
  int j = t;
  float w2j = fw2[j];
  float b1j = fb1[j];
  for (int g = 0; g < NUM_GRAPHS; ++g) {
    float acc = b1j;
    for (int k = 0; k < 128; ++k) acc += ge[g * 128 + k] * fw1[k * 128 + j];
    acc = fmaxf(acc, 0.f);
    atomicAdd(&out[g], acc * w2j);
  }
  if (t == 0) {
    float b2 = fb2[0];
    for (int g = 0; g < NUM_GRAPHS; ++g) atomicAdd(&out[g], b2);
  }
}

// ---------------------------------------------------------------------------
// Host launcher
// ---------------------------------------------------------------------------
extern "C" void kernel_launch(void* const* d_in, const int* in_sizes, int n_in,
                              void* d_out, int out_size, void* d_ws,
                              size_t ws_size, hipStream_t stream) {
  const float* x         = (const float*)d_in[0];
  const float* edge_attr = (const float*)d_in[1];
  const int*   edge_index= (const int*)d_in[2];
  const int*   batch     = (const int*)d_in[3];
  const float* node_w1 = (const float*)d_in[4];
  const float* node_b1 = (const float*)d_in[5];
  const float* node_w2 = (const float*)d_in[6];
  const float* node_b2 = (const float*)d_in[7];
  const float* dist_w1 = (const float*)d_in[8];
  const float* dist_b1 = (const float*)d_in[9];
  const float* dist_w2 = (const float*)d_in[10];
  const float* dist_b2 = (const float*)d_in[11];
  const float* ang_w1  = (const float*)d_in[12];
  const float* ang_b1  = (const float*)d_in[13];
  const float* ang_w2  = (const float*)d_in[14];
  const float* ang_b2  = (const float*)d_in[15];
  const float* raw_w1  = (const float*)d_in[16];
  const float* raw_b1  = (const float*)d_in[17];
  const float* raw_w2  = (const float*)d_in[18];
  const float* raw_b2  = (const float*)d_in[19];
  const float* e2n_w   = (const float*)d_in[20];
  const float* e2n_b   = (const float*)d_in[21];
  const float* gin_w1  = (const float*)d_in[22];
  const float* gin_b1  = (const float*)d_in[23];
  const float* gin_w2  = (const float*)d_in[24];
  const float* gin_b2  = (const float*)d_in[25];
  const float* fc_w1   = (const float*)d_in[26];
  const float* fc_b1   = (const float*)d_in[27];
  const float* fc_w2   = (const float*)d_in[28];
  const float* fc_b2   = (const float*)d_in[29];

  int N = in_sizes[0] / 16;
  int E = in_sizes[1] / 8;

  char* ws = (char*)d_ws;
  size_t off = 0;
  auto wsalloc = [&](size_t bytes) {
    size_t o = off;
    off += (bytes + 255) & ~(size_t)255;
    return o;
  };
  auto wbytes = [](int K, int Nn) {
    size_t kc = (size_t)(K + 31) / 32;
    return kc * 32 * (size_t)Nn * sizeof(__bf16);
  };

  size_t o_enh    = wsalloc((size_t)N * 192 * 4);
  size_t o_gin    = wsalloc((size_t)N * 192 * 4);
  size_t o_sums   = wsalloc((size_t)NUM_GRAPHS * 128 * 4);
  size_t o_counts = wsalloc((size_t)NUM_GRAPHS * 4);
  size_t o_wn1 = wsalloc(wbytes(16, 128));
  size_t o_wn2 = wsalloc(wbytes(128, 128));
  size_t o_wd2 = wsalloc(wbytes(32, 32));
  size_t o_wa2 = wsalloc(wbytes(32, 32));
  size_t o_wr2 = wsalloc(wbytes(32, 32));
  size_t o_we  = wsalloc(wbytes(96, 64));
  size_t o_wg1 = wsalloc(wbytes(192, 128));
  size_t o_wg2 = wsalloc(wbytes(128, 128));

  float*  enh    = (float*)(ws + o_enh);
  float*  ginb   = (float*)(ws + o_gin);
  float*  sums   = (float*)(ws + o_sums);
  float*  counts = (float*)(ws + o_counts);
  __bf16* wn1 = (__bf16*)(ws + o_wn1);
  __bf16* wn2 = (__bf16*)(ws + o_wn2);
  __bf16* wd2 = (__bf16*)(ws + o_wd2);
  __bf16* wa2 = (__bf16*)(ws + o_wa2);
  __bf16* wr2 = (__bf16*)(ws + o_wr2);
  __bf16* we  = (__bf16*)(ws + o_we);
  __bf16* wg1 = (__bf16*)(ws + o_wg1);
  __bf16* wg2 = (__bf16*)(ws + o_wg2);

  hipMemsetAsync(enh, 0, (size_t)N * 192 * 4, stream);
  hipMemsetAsync(sums, 0, (size_t)NUM_GRAPHS * 128 * 4, stream);
  hipMemsetAsync(counts, 0, (size_t)NUM_GRAPHS * 4, stream);
  hipMemsetAsync(d_out, 0, (size_t)out_size * 4, stream);

  auto swz = [&](const float* W, __bf16* dst, int K, int Nn) {
    int total = ((K + 31) / 32) * (Nn / 16) * 32;
    swizzle_weight_kernel<<<(total + 127) / 128, 128, 0, stream>>>(W, dst, K, Nn);
  };
  swz(node_w1, wn1, 16, 128);
  swz(node_w2, wn2, 128, 128);
  swz(dist_w2, wd2, 32, 32);
  swz(ang_w2,  wa2, 32, 32);
  swz(raw_w2,  wr2, 32, 32);
  swz(e2n_w,   we,  96, 64);
  swz(gin_w1,  wg1, 192, 128);
  swz(gin_w2,  wg2, 128, 128);

  node_mlp_kernel<<<(N + 63) / 64, 128, 0, stream>>>(x, node_b1, node_b2, wn1,
                                                     wn2, enh, N);

  edge_feature_kernel<<<(E + 127) / 128, 256, 0, stream>>>(
      x, edge_attr, edge_index, dist_w1, dist_b1, dist_b2, ang_w1, ang_b1,
      ang_b2, raw_w1, raw_b1, raw_b2, wd2, wa2, wr2, we, e2n_b, enh, E);

  hipMemcpyAsync(ginb, enh, (size_t)N * 192 * 4, hipMemcpyDeviceToDevice,
                 stream);

  long tot = (long)E * 48;
  scatter_neigh_kernel<<<(int)((tot + 255) / 256), 256, 0, stream>>>(
      enh, ginb, edge_index, E);

  counts_kernel<<<(N + 255) / 256, 256, 0, stream>>>(batch, counts, N);

  gin_mlp_kernel<<<(N + 63) / 64, 128, 0, stream>>>(ginb, gin_b1, gin_b2, wg1,
                                                    wg2, batch, sums, N);

  final_kernel<<<1, 128, 0, stream>>>(sums, counts, fc_w1, fc_b1, fc_w2, fc_b2,
                                      (float*)d_out);
}
